// CrossTransformerBlock2D_13013750907333
// MI455X (gfx1250) — matrix operand order repaired and checked by reading, not compiled, not run on hardware
//
#include <hip/hip_runtime.h>

// ---------------- types ----------------
typedef __attribute__((ext_vector_type(16))) __bf16 v16bf;
typedef __attribute__((ext_vector_type(8)))  __bf16 v8bf;
typedef __attribute__((ext_vector_type(8)))  float  v8f;

#define BATCH   8
#define CDIM    256
#define NPIX    4096      // 64*64
#define NHEADS  8
#define CH      32
#define HID     680
#define HIDP    704       // HID padded to /32 (K of ffn_out)
#define HID2    1360
#define HID2P   1408      // 2*HID padded to /64 (O of ffn_in)
#define KSPLIT  8         // attention qk^T K-dim split
#define PSPLIT  8         // attention attn*v pixel split

static __device__ __forceinline__ unsigned short f2bf(float f) {
    unsigned int u = __builtin_bit_cast(unsigned int, f);
    unsigned int r = u + 0x7FFFu + ((u >> 16) & 1u);   // RNE
    return (unsigned short)(r >> 16);
}
static __device__ __forceinline__ float bf2f(unsigned short h) {
    unsigned int u = ((unsigned int)h) << 16;
    return __builtin_bit_cast(float, u);
}
static __device__ __forceinline__ v16bf cat8(v8bf lo, v8bf hi) {
    return __builtin_shufflevector(lo, hi, 0,1,2,3,4,5,6,7,8,9,10,11,12,13,14,15);
}

// ---------------- weight convert (f32 -> bf16, zero-padded) ----------------
__global__ void convert_w_kernel(const float* __restrict__ src, unsigned short* __restrict__ dst,
                                 int O, int K, int Opad, int Kpad) {
    int idx = blockIdx.x * 256 + threadIdx.x;
    int tot = Opad * Kpad;
    if (idx >= tot) return;
    int o = idx / Kpad, kk = idx - o * Kpad;
    float v = (o < O && kk < K) ? src[(size_t)o * K + kk] : 0.f;
    dst[idx] = f2bf(v);
}

// ---------------- channel LayerNorm: f32 [B,C,N] -> bf16 [B,C,N] ----------------
__global__ __launch_bounds__(256) void ln_kernel(const float* __restrict__ src,
                                                 const float* __restrict__ w,
                                                 const float* __restrict__ bias,
                                                 unsigned short* __restrict__ dst) {
    int bid = blockIdx.x;             // 128 blocks: b = bid/16, pixel tile = bid%16
    int b = bid >> 4;
    int p = ((bid & 15) << 8) + threadIdx.x;
    const float* s = src + (size_t)b * CDIM * NPIX + p;
    float sum = 0.f, ss = 0.f;
    for (int c = 0; c < CDIM; ++c) { float v = s[(size_t)c * NPIX]; sum += v; ss += v * v; }
    float mu = sum * (1.f / CDIM);
    float var = ss * (1.f / CDIM) - mu * mu;
    float rs = rsqrtf(var + 1e-5f);
    unsigned short* d = dst + (size_t)b * CDIM * NPIX + p;
    for (int c = 0; c < CDIM; ++c) {
        float v = s[(size_t)c * NPIX];
        d[(size_t)c * NPIX] = f2bf((v - mu) * rs * w[c] + bias[c]);
    }
}

// ---------------- depthwise 3x3 for qkv + sumsq(q),sumsq(k) ----------------
// grid (3C, B), block 256.  ssq layout: [0..B*C) = q, [B*C..2*B*C) = k
__global__ __launch_bounds__(256) void dwconv_qkv_kernel(const unsigned short* __restrict__ xn,
                                                         const unsigned short* __restrict__ yn,
                                                         const float* __restrict__ dw,
                                                         unsigned short* __restrict__ q,
                                                         unsigned short* __restrict__ k,
                                                         unsigned short* __restrict__ v,
                                                         float* __restrict__ ssq) {
    int cg = blockIdx.x, b = blockIdx.y, tid = threadIdx.x;
    int ic = cg < CDIM ? cg : (cg < 2 * CDIM ? cg - CDIM : cg - 2 * CDIM);
    const unsigned short* src = (cg < CDIM ? xn : yn) + ((size_t)b * CDIM + ic) * NPIX;
    unsigned short* dst; float* sp = nullptr;
    if (cg < CDIM)            { dst = q + ((size_t)b * CDIM + cg) * NPIX;           sp = ssq + b * CDIM + cg; }
    else if (cg < 2 * CDIM)   { dst = k + ((size_t)b * CDIM + cg - CDIM) * NPIX;    sp = ssq + BATCH * CDIM + b * CDIM + (cg - CDIM); }
    else                      { dst = v + ((size_t)b * CDIM + cg - 2 * CDIM) * NPIX; }
    float wt[9];
#pragma unroll
    for (int i = 0; i < 9; ++i) wt[i] = dw[cg * 9 + i];
    float loc = 0.f;
    for (int it = 0; it < 16; ++it) {
        int p = it * 256 + tid;
        int y0 = p >> 6, x0 = p & 63;
        float acc = 0.f;
#pragma unroll
        for (int dy = -1; dy <= 1; ++dy) {
            int yy = y0 + dy;
            if ((unsigned)yy < 64u) {
#pragma unroll
                for (int dx = -1; dx <= 1; ++dx) {
                    int xx = x0 + dx;
                    if ((unsigned)xx < 64u)
                        acc += wt[(dy + 1) * 3 + (dx + 1)] * bf2f(src[yy * 64 + xx]);
                }
            }
        }
        dst[p] = f2bf(acc);
        loc += acc * acc;
    }
    __shared__ float red[256];
    red[tid] = loc; __syncthreads();
    for (int s = 128; s > 0; s >>= 1) { if (tid < s) red[tid] += red[tid + s]; __syncthreads(); }
    if (tid == 0 && sp) *sp = red[0];   // block covers whole channel -> exact, no atomics
}

// ---------------- attention 1/3: partial S = q k^T over a K chunk ----------------
// grid (NHEADS, BATCH, KSPLIT), 128 threads (4 waves, 2x2 16x16 tiles).
// spart layout: [kc][b*NHEADS+h][32][32] f32
__global__ __launch_bounds__(128) void attn_qk_kernel(const unsigned short* __restrict__ q,
                                                      const unsigned short* __restrict__ kk,
                                                      float* __restrict__ spart) {
    int h = blockIdx.x, b = blockIdx.y, kc = blockIdx.z, tid = threadIdx.x;
    int l = tid & 31, lg = l >> 4, li = l & 15;
    int w = tid >> 5, mc = w & 1, nd = w >> 1;
    size_t hb = ((size_t)b * CDIM + h * CH) * NPIX;
    const int klen = NPIX / KSPLIT;              // 512
    const int kbeg = kc * klen;

    const unsigned short* ar = q  + hb + (size_t)(mc * 16 + li) * NPIX + kbeg;
    const unsigned short* br = kk + hb + (size_t)(nd * 16 + li) * NPIX + kbeg;
    v8f acc = {0.f, 0.f, 0.f, 0.f, 0.f, 0.f, 0.f, 0.f};
    for (int pk = 0; pk < klen; pk += 32) {
        v8bf lo = *(const v8bf*)(ar + pk + lg * 8);
        v8bf hi = *(const v8bf*)(ar + pk + 16 + lg * 8);
        v16bf a = cat8(lo, hi);
        v16bf bm = cat8(*(const v8bf*)(br + pk + lg * 8), *(const v8bf*)(br + pk + 16 + lg * 8));
        acc = __builtin_amdgcn_wmma_f32_16x16x32_bf16(false, a, false, bm, (short)0, acc, false, false);
    }
    float* out = spart + ((size_t)(kc * BATCH * NHEADS) + b * NHEADS + h) * 1024;
#pragma unroll
    for (int j = 0; j < 8; ++j) {
        int c = mc * 16 + j + 8 * lg;
        int d = nd * 16 + li;
        out[c * 32 + d] = acc[j];
    }
}

// ---------------- attention 2/3: reduce partials, scale, softmax -> bf16 attn ----------------
// grid (BATCH*NHEADS), 128 threads.
__global__ __launch_bounds__(128) void attn_softmax_kernel(const float* __restrict__ spart,
                                                           const float* __restrict__ ssq,
                                                           const float* __restrict__ temp,
                                                           unsigned short* __restrict__ attn_g) {
    __shared__ float S[32][33];
    __shared__ float inq[32], ink[32];
    int bh = blockIdx.x, tid = threadIdx.x;
    int b = bh / NHEADS, h = bh - b * NHEADS;

    if (tid < 32)       inq[tid]      = 1.f / fmaxf(sqrtf(ssq[b * CDIM + h * CH + tid]), 1e-12f);
    else if (tid < 64)  ink[tid - 32] = 1.f / fmaxf(sqrtf(ssq[BATCH * CDIM + b * CDIM + h * CH + (tid - 32)]), 1e-12f);
    __syncthreads();

    float tm = temp[h];
#pragma unroll
    for (int i = 0; i < 8; ++i) {
        int e = tid + i * 128;
        float s = 0.f;
#pragma unroll
        for (int kc = 0; kc < KSPLIT; ++kc)
            s += spart[((size_t)(kc * BATCH * NHEADS) + bh) * 1024 + e];
        int c = e >> 5, d = e & 31;
        S[c][d] = s * tm * inq[c] * ink[d];
    }
    __syncthreads();
    if (tid < 32) {
        float m = -1e30f;
#pragma unroll
        for (int d = 0; d < 32; ++d) m = fmaxf(m, S[tid][d]);
        float e[32]; float sum = 0.f;
#pragma unroll
        for (int d = 0; d < 32; ++d) { e[d] = expf(S[tid][d] - m); sum += e[d]; }
        float inv = 1.f / sum;
#pragma unroll
        for (int d = 0; d < 32; ++d)
            attn_g[(size_t)bh * 1024 + tid * 32 + d] = f2bf(e[d] * inv);
    }
}

// ---------------- attention 3/3: out[c,p] = sum_d attn[c,d] * v[d,p] ----------------
// grid (NHEADS, BATCH, PSPLIT), 128 threads (4 waves).
__global__ __launch_bounds__(128) void attn_av_kernel(const unsigned short* __restrict__ attn_g,
                                                      const unsigned short* __restrict__ vv,
                                                      unsigned short* __restrict__ ao) {
    __shared__ alignas(32) unsigned short vt[32][48];
    int h = blockIdx.x, b = blockIdx.y, pc = blockIdx.z, tid = threadIdx.x;
    int l = tid & 31, lg = l >> 4, li = l & 15;
    int w = tid >> 5, mc = w & 1, np = w >> 1;
    size_t hb = ((size_t)b * CDIM + h * CH) * NPIX;
    const unsigned short* ab = attn_g + (size_t)(b * NHEADS + h) * 1024;

    // A operand: attn rows (constant over pixel loop)
    int c = mc * 16 + li;
    v8bf alo = *(const v8bf*)(ab + c * 32 + lg * 8);
    v8bf ahi = *(const v8bf*)(ab + c * 32 + 16 + lg * 8);
    v16bf aA = cat8(alo, ahi);
    const v8f zero = {0.f, 0.f, 0.f, 0.f, 0.f, 0.f, 0.f, 0.f};

    const int plen = NPIX / PSPLIT;          // 512
    for (int it = 0; it < plen / 32; ++it) { // 16 iters of 32 pixels
        int p0 = pc * plen + it * 32;
        int dr = tid >> 2, pj = (tid & 3) * 8;
        const unsigned short* sp = vv + hb + (size_t)dr * NPIX + p0 + pj;
#pragma unroll
        for (int s2 = 0; s2 < 8; ++s2) vt[pj + s2][dr] = sp[s2];   // transpose v tile
        __syncthreads();
        v16bf bm = cat8(*(const v8bf*)(&vt[np * 16 + li][lg * 8]), *(const v8bf*)(&vt[np * 16 + li][16 + lg * 8]));
        v8f o = __builtin_amdgcn_wmma_f32_16x16x32_bf16(false, aA, false, bm, (short)0, zero, false, false);
        int p = p0 + np * 16 + li;
#pragma unroll
        for (int j = 0; j < 8; ++j)
            ao[hb + (size_t)(mc * 16 + j + 8 * lg) * NPIX + p] = f2bf(o[j]);
        __syncthreads();
    }
}

// ---------------- generic channel-major WMMA GEMM: out[o,p] = W[o,:] . act[:,p] ----------------
// A = bf16 weights [Opad,Kpad] row-major (direct global), B = act tile transposed via LDS.
// grid (16, Opad/64, B), block 256 (8 waves: 2 M x 4 N), wave computes 32 x 64.
__global__ __launch_bounds__(256) void gemm_cm_kernel(const unsigned short* __restrict__ W,
                                                      const unsigned short* __restrict__ act,
                                                      void* __restrict__ out,
                                                      const float* __restrict__ res,
                                                      int Kpad, int Opad, int outf32) {
    __shared__ alignas(32) unsigned short sm[256 * 48];
    int tid = threadIdx.x;
    int p0 = blockIdx.x * 256, o0 = blockIdx.y * 64, b = blockIdx.z;
    int l = tid & 31, lg = l >> 4, li = l & 15;
    int wm = (tid >> 5) & 1, wn = tid >> 6;
    v8f z = {0.f, 0.f, 0.f, 0.f, 0.f, 0.f, 0.f, 0.f};
    v8f acc0[4] = {z, z, z, z};
    v8f acc1[4] = {z, z, z, z};
    const size_t actb = (size_t)b * Kpad * NPIX;

    for (int kb = 0; kb < Kpad; kb += 32) {
        __syncthreads();
        // stage act[kb..kb+31][p0..p0+255] into LDS transposed -> sm[p][k]
#pragma unroll
        for (int i = 0; i < 8; ++i) {
            int cid = tid + i * 256;
            int kl = cid >> 6, pj = (cid & 63) << 2;
            const unsigned short* sp = act + actb + (size_t)(kb + kl) * NPIX + p0 + pj;
            unsigned short a0 = sp[0], a1 = sp[1], a2 = sp[2], a3 = sp[3];
            sm[(pj + 0) * 48 + kl] = a0;
            sm[(pj + 1) * 48 + kl] = a1;
            sm[(pj + 2) * 48 + kl] = a2;
            sm[(pj + 3) * 48 + kl] = a3;
        }
        __syncthreads();
        const unsigned short* wr0 = W + (size_t)(o0 + wm * 32 + li) * Kpad + kb + lg * 8;
        const unsigned short* wr1 = wr0 + (size_t)16 * Kpad;
        v16bf a0 = cat8(*(const v8bf*)wr0, *(const v8bf*)(wr0 + 16));
        v16bf a1 = cat8(*(const v8bf*)wr1, *(const v8bf*)(wr1 + 16));
#pragma unroll
        for (int nt = 0; nt < 4; ++nt) {
            const unsigned short* bp = &sm[(wn * 64 + nt * 16 + li) * 48 + lg * 8];
            v16bf bm = cat8(*(const v8bf*)bp, *(const v8bf*)(bp + 16));
            acc0[nt] = __builtin_amdgcn_wmma_f32_16x16x32_bf16(false, a0, false, bm, (short)0, acc0[nt], false, false);
            acc1[nt] = __builtin_amdgcn_wmma_f32_16x16x32_bf16(false, a1, false, bm, (short)0, acc1[nt], false, false);
        }
    }
#pragma unroll
    for (int half = 0; half < 2; ++half) {
#pragma unroll
        for (int nt = 0; nt < 4; ++nt) {
            int p = p0 + wn * 64 + nt * 16 + li;
#pragma unroll
            for (int j = 0; j < 8; ++j) {
                int o = o0 + wm * 32 + half * 16 + j + 8 * lg;
                float vvv = half ? acc1[nt][j] : acc0[nt][j];
                if (res) vvv += res[((size_t)b * CDIM + o) * NPIX + p];
                size_t idx = ((size_t)b * Opad + o) * NPIX + p;
                if (outf32) ((float*)out)[idx] = vvv;
                else        ((unsigned short*)out)[idx] = f2bf(vvv);
            }
        }
    }
}

// ---------------- FFN depthwise 3x3 + gated exact GELU ----------------
// grid (HIDP, B): cg < HID computes gelu(dw(t1))*dw(t2); pad rows write zero.
__global__ __launch_bounds__(256) void ffn_dwgelu_kernel(const unsigned short* __restrict__ t,
                                                         const float* __restrict__ dw,
                                                         unsigned short* __restrict__ g) {
    int cg = blockIdx.x, b = blockIdx.y, tid = threadIdx.x;
    unsigned short* gd = g + ((size_t)b * HIDP + cg) * NPIX;
    if (cg >= HID) {
        for (int it = 0; it < 16; ++it) gd[it * 256 + tid] = 0;
        return;
    }
    const unsigned short* s1 = t + ((size_t)b * HID2P + cg) * NPIX;
    const unsigned short* s2 = t + ((size_t)b * HID2P + HID + cg) * NPIX;
    float w1[9], w2[9];
#pragma unroll
    for (int i = 0; i < 9; ++i) { w1[i] = dw[cg * 9 + i]; w2[i] = dw[(HID + cg) * 9 + i]; }
    for (int it = 0; it < 16; ++it) {
        int p = it * 256 + tid;
        int y0 = p >> 6, x0 = p & 63;
        float a1 = 0.f, a2 = 0.f;
#pragma unroll
        for (int dy = -1; dy <= 1; ++dy) {
            int yy = y0 + dy;
            if ((unsigned)yy < 64u) {
#pragma unroll
                for (int dx = -1; dx <= 1; ++dx) {
                    int xx = x0 + dx;
                    if ((unsigned)xx < 64u) {
                        int wi = (dy + 1) * 3 + (dx + 1);
                        int np = yy * 64 + xx;
                        a1 += w1[wi] * bf2f(s1[np]);
                        a2 += w2[wi] * bf2f(s2[np]);
                    }
                }
            }
        }
        float ge = 0.5f * a1 * (1.f + erff(a1 * 0.70710678f));   // exact GELU
        gd[p] = f2bf(ge * a2);
    }
}

// ---------------- launch ----------------
extern "C" void kernel_launch(void* const* d_in, const int* in_sizes, int n_in,
                              void* d_out, int out_size, void* d_ws, size_t ws_size,
                              hipStream_t stream) {
    (void)in_sizes; (void)n_in; (void)out_size; (void)ws_size;
    const float* x        = (const float*)d_in[0];
    const float* y        = (const float*)d_in[1];
    const float* ln1x_w   = (const float*)d_in[2];
    const float* ln1x_b   = (const float*)d_in[3];
    const float* ln1y_w   = (const float*)d_in[4];
    const float* ln1y_b   = (const float*)d_in[5];
    const float* ln2_w    = (const float*)d_in[6];
    const float* ln2_b    = (const float*)d_in[7];
    const float* temp     = (const float*)d_in[8];
    const float* qkv_dw   = (const float*)d_in[9];
    const float* proj_w   = (const float*)d_in[10];
    const float* ffn_in_w = (const float*)d_in[11];
    const float* ffn_dw   = (const float*)d_in[12];
    const float* ffn_out_w= (const float*)d_in[13];

    char* wsb = (char*)d_ws;
    const size_t SB2 = (size_t)BATCH * CDIM * NPIX * 2;   // one bf16 [B,C,N] buffer
    unsigned short* xn  = (unsigned short*)(wsb + 0 * SB2);
    unsigned short* yn  = (unsigned short*)(wsb + 1 * SB2);
    unsigned short* qb  = (unsigned short*)(wsb + 2 * SB2);
    unsigned short* kb  = (unsigned short*)(wsb + 3 * SB2);
    unsigned short* vb  = (unsigned short*)(wsb + 4 * SB2);
    unsigned short* ao  = yn;   // yn dead after dwconv_qkv
    unsigned short* ln2 = xn;   // xn dead after dwconv_qkv
    unsigned short* g   = qb;   // q/k/v dead after attention (uses 3 buffers' worth)
    float*          xa  = (float*)(wsb + 5 * SB2);
    size_t off = 5 * SB2 + (size_t)BATCH * CDIM * NPIX * 4;
    unsigned short* t     = (unsigned short*)(wsb + off); off += (size_t)BATCH * HID2P * NPIX * 2;
    unsigned short* wproj = (unsigned short*)(wsb + off); off += (size_t)CDIM * CDIM * 2;
    unsigned short* win   = (unsigned short*)(wsb + off); off += (size_t)HID2P * CDIM * 2;
    unsigned short* wout  = (unsigned short*)(wsb + off); off += (size_t)CDIM * HIDP * 2;
    float* ssq            = (float*)(wsb + off);          off += (size_t)2 * BATCH * CDIM * 4;
    float* spart          = (float*)(wsb + off);          off += (size_t)KSPLIT * BATCH * NHEADS * 1024 * 4;
    unsigned short* attn_g= (unsigned short*)(wsb + off); off += (size_t)BATCH * NHEADS * 1024 * 2;

    // weights -> bf16 (zero padded)
    convert_w_kernel<<<(CDIM * CDIM + 255) / 256, 256, 0, stream>>>(proj_w, wproj, CDIM, CDIM, CDIM, CDIM);
    convert_w_kernel<<<(HID2P * CDIM + 255) / 256, 256, 0, stream>>>(ffn_in_w, win, HID2, CDIM, HID2P, CDIM);
    convert_w_kernel<<<(CDIM * HIDP + 255) / 256, 256, 0, stream>>>(ffn_out_w, wout, CDIM, HID, CDIM, HIDP);

    // LayerNorms
    ln_kernel<<<128, 256, 0, stream>>>(x, ln1x_w, ln1x_b, xn);
    ln_kernel<<<128, 256, 0, stream>>>(y, ln1y_w, ln1y_b, yn);

    // depthwise qkv + q/k sum-of-squares
    dwconv_qkv_kernel<<<dim3(3 * CDIM, BATCH), 256, 0, stream>>>(xn, yn, qkv_dw, qb, kb, vb, ssq);

    // attention: partial qk^T -> softmax -> attn*v
    attn_qk_kernel<<<dim3(NHEADS, BATCH, KSPLIT), 128, 0, stream>>>(qb, kb, spart);
    attn_softmax_kernel<<<BATCH * NHEADS, 128, 0, stream>>>(spart, ssq, temp, attn_g);
    attn_av_kernel<<<dim3(NHEADS, BATCH, PSPLIT), 128, 0, stream>>>(attn_g, vb, ao);

    // x = x + proj(ao)            (f32 out)
    gemm_cm_kernel<<<dim3(16, CDIM / 64, BATCH), 256, 0, stream>>>(wproj, ao, (void*)xa, x, CDIM, CDIM, 1);

    // LN2
    ln_kernel<<<128, 256, 0, stream>>>(xa, ln2_w, ln2_b, ln2);

    // t = Win . ln2               (bf16 out, 1408 rows)
    gemm_cm_kernel<<<dim3(16, HID2P / 64, BATCH), 256, 0, stream>>>(win, ln2, (void*)t, nullptr, CDIM, HID2P, 0);

    // g = gelu(dw(t1)) * dw(t2)   (704 rows, pad rows zeroed)
    ffn_dwgelu_kernel<<<dim3(HIDP, BATCH), 256, 0, stream>>>(t, ffn_dw, g);

    // out = xa + Wout . g         (f32 out -> d_out)
    gemm_cm_kernel<<<dim3(16, CDIM / 64, BATCH), 256, 0, stream>>>(wout, g, d_out, xa, HIDP, CDIM, 1);
}
